// MultidimensionalLSTM_16063177687694
// MI455X (gfx1250) — matrix operand order repaired and checked by reading, not compiled, not run on hardware
//
#include <hip/hip_runtime.h>
#include <math.h>

// Problem constants
#define BB   4096   // batch
#define TT   64     // timesteps
#define FF   89     // input features
#define HH   1024   // hidden
#define OO   89     // output
#define NN   4096   // 4*H gate width
#define KX   96     // padded x-feature section of K (89 -> 96)
#define KPAD 1120   // KX + HH, = 35 * 32
#define KSTEPS (KPAD / 32)
#define LDS_ROW 40  // padded LDS row length in shorts (20 dwords -> conflict-free frag reads)

typedef __attribute__((ext_vector_type(16))) __bf16 v16bf;
typedef __attribute__((ext_vector_type(8)))  float  v8f;
typedef int v4i __attribute__((vector_size(16)));   // int4 payload type for async LDS copies

union Frag16 { v16bf v; unsigned u[8]; };

// ---- gfx1250 async global->LDS path (upstream clang builtins), with fallback ----
#if defined(__has_builtin)
#if __has_builtin(__builtin_amdgcn_global_load_async_to_lds_b128)
#define HAS_ASYNC_LDS 1
#endif
#endif

// Builtin signature (from compiler diagnostic): (v4i as1*, v4i as3*, imm offset, imm cpol)
#define AS1G(p) ((__attribute__((address_space(1))) v4i*)(p))
#define AS3L(p) ((__attribute__((address_space(3))) v4i*)(p))

static __device__ __forceinline__ void wait_async_zero() {
#if defined(HAS_ASYNC_LDS)
#if __has_builtin(__builtin_amdgcn_s_wait_asynccnt)
  __builtin_amdgcn_s_wait_asynccnt(0);
#else
  asm volatile("s_wait_asynccnt 0x0" ::: "memory");
#endif
#endif
}

static __device__ __forceinline__ unsigned short f2bf(float f) {
  union { float f; unsigned u; } v; v.f = f;
  unsigned r = v.u + 0x7FFFu + ((v.u >> 16) & 1u);   // round-to-nearest-even
  return (unsigned short)(r >> 16);
}
static __device__ __forceinline__ float bf2f(unsigned short h) {
  union { unsigned u; float f; } v; v.u = ((unsigned)h) << 16; return v.f;
}

// ---------------------------------------------------------------------------
// Build combined transposed bf16 weight: Wct[n][k], k-major.
//   k in [0,89)      -> Wk[k][n]
//   k in [89,96)     -> 0 (pad)
//   k in [96,1120)   -> Wr[k-96][n]
// ---------------------------------------------------------------------------
__global__ __launch_bounds__(256) void build_wct_kernel(
    const float* __restrict__ Wk, const float* __restrict__ Wr,
    unsigned short* __restrict__ wct) {
  size_t i = (size_t)blockIdx.x * 256 + threadIdx.x;
  if (i >= (size_t)NN * KPAD) return;
  int n = (int)(i / KPAD);
  int k = (int)(i % KPAD);
  float v = 0.0f;
  if (k < FF)       v = Wk[(size_t)k * NN + n];
  else if (k >= KX) v = Wr[(size_t)(k - KX) * NN + n];
  wct[i] = f2bf(v);
}

// Zero h (bf16) and c (f32)
__global__ __launch_bounds__(256) void init_state_kernel(
    unsigned short* __restrict__ h, float* __restrict__ c) {
  size_t i = (size_t)blockIdx.x * 256 + threadIdx.x;
  if (i >= (size_t)BB * HH) return;
  h[i] = 0;
  c[i] = 0.0f;
}

// ---------------------------------------------------------------------------
// One recurrence GEMM step:  Z[b][n] = sum_k A[b][k] * Wct[n][k]
// A[b][k] = x[b][t][k] (k<89, bf16-converted, zero-padded to 96) or h[b][k-96]
// 128x128 block tile, 8 waves, each wave 64x32 via 4x2 WMMA 16x16x32 bf16 frags.
// Double-buffered LDS staging; async global->LDS prefetch of slab k+1 overlaps
// the WMMA work on slab k.
// ---------------------------------------------------------------------------
__global__ __launch_bounds__(256) void lstm_gemm_step_kernel(
    const float* __restrict__ x,            // [B,T,F] f32
    const unsigned short* __restrict__ wct, // [NN][KPAD] bf16
    const unsigned short* __restrict__ h,   // [B][H] bf16
    float* __restrict__ z,                  // [B][NN] f32
    int t) {
  __shared__ unsigned short As[2][128 * LDS_ROW];
  __shared__ unsigned short Bs[2][128 * LDS_ROW];

  const int tid  = threadIdx.x;
  const int lane = tid & 31;
  const int wave = tid >> 5;
  const int tm = blockIdx.x * 128;
  const int tn = blockIdx.y * 128;
  const int wm = (wave >> 2) * 64;   // 0 or 64
  const int wn = (wave & 3) * 32;    // 0,32,64,96

  v8f acc[4][2];
#pragma unroll
  for (int i = 0; i < 4; ++i)
#pragma unroll
    for (int j = 0; j < 2; ++j) acc[i][j] = (v8f){0,0,0,0,0,0,0,0};

  const int row2 = tid >> 1;          // 0..127 : row staged by this thread
  const int seg  = (tid & 1) * 16;    // 0 or 16: which half of the 32-wide K slab

  const int hi   = (lane < 16) ? 0 : 1;
  const int l15  = lane & 15;

  // Stage the 32-wide K slab `kk` into LDS buffer `buf` (this thread's 32B of A + 32B of B)
  auto stage = [&](int kk, int buf) {
    const int k0 = kk * 32;
    // ---- B tile: Wct rows tn..tn+127 (n-major, k contiguous) ----
    {
      const unsigned short* src = wct + (size_t)(tn + row2) * KPAD + k0 + seg;
      unsigned short* dst = &Bs[buf][row2 * LDS_ROW + seg];
#if defined(HAS_ASYNC_LDS)
      __builtin_amdgcn_global_load_async_to_lds_b128(AS1G(src), AS3L(dst), 0, 0);
      __builtin_amdgcn_global_load_async_to_lds_b128(AS1G(src), AS3L(dst), 16, 0);
#else
      uint4 b0 = ((const uint4*)src)[0];
      uint4 b1 = ((const uint4*)src)[1];
      ((uint4*)dst)[0] = b0; ((uint4*)dst)[1] = b1;
#endif
    }
    // ---- A tile: rows tm..tm+127 of concat(x_t_pad, h) ----
    if (k0 >= KX) {
      const unsigned short* src = h + (size_t)(tm + row2) * HH + (k0 - KX) + seg;
      unsigned short* dst = &As[buf][row2 * LDS_ROW + seg];
#if defined(HAS_ASYNC_LDS)
      __builtin_amdgcn_global_load_async_to_lds_b128(AS1G(src), AS3L(dst), 0, 0);
      __builtin_amdgcn_global_load_async_to_lds_b128(AS1G(src), AS3L(dst), 16, 0);
#else
      uint4 a0 = ((const uint4*)src)[0];
      uint4 a1 = ((const uint4*)src)[1];
      ((uint4*)dst)[0] = a0; ((uint4*)dst)[1] = a1;
#endif
    } else {
      const float* xs = x + ((size_t)(tm + row2) * TT + t) * FF;
#pragma unroll
      for (int j = 0; j < 16; ++j) {
        int kc = k0 + seg + j;
        As[buf][row2 * LDS_ROW + seg + j] = (kc < FF) ? f2bf(xs[kc]) : (unsigned short)0;
      }
    }
  };

  // Prologue: fill buffer 0 with slab 0
  stage(0, 0);
  wait_async_zero();
  __syncthreads();

  for (int kk = 0; kk < KSTEPS; ++kk) {
    const int cur = kk & 1;
    // Prefetch next slab into the other buffer while we compute on this one
    if (kk + 1 < KSTEPS) stage(kk + 1, cur ^ 1);

    const unsigned* Aw = (const unsigned*)As[cur];   // [row][LDS_ROW/2] dwords (k-pairs)
    const unsigned* Bw = (const unsigned*)Bs[cur];

    // B fragments: lane<16 -> N=wn+j*16+lane, k-pairs 0..7 ; lane>=16 -> k-pairs 8..15
    Frag16 bf[2];
#pragma unroll
    for (int j = 0; j < 2; ++j) {
      const int n = wn + j * 16 + l15;
      const int base = n * (LDS_ROW / 2) + hi * 8;
#pragma unroll
      for (int e = 0; e < 8; ++e) bf[j].u[e] = Bw[base + e];
    }
    // A fragments: lane<16 -> M=m0+lane, k-pairs 0..7 ; lane>=16 -> k-pairs 8..15
#pragma unroll
    for (int i = 0; i < 4; ++i) {
      const int m = wm + i * 16 + l15;
      const int abase = m * (LDS_ROW / 2) + hi * 8;
      Frag16 af;
#pragma unroll
      for (int e = 0; e < 4; ++e) af.u[e]     = Aw[abase + e];
#pragma unroll
      for (int e = 0; e < 4; ++e) af.u[4 + e] = Aw[abase + 4 + e];
#pragma unroll
      for (int j = 0; j < 2; ++j)
        acc[i][j] = __builtin_amdgcn_wmma_f32_16x16x32_bf16(
            false, af.v, false, bf[j].v, (short)0, acc[i][j], false, false);
    }

    // My async prefetch (if any) has landed; barrier also covers ds stores of the
    // x-conversion path and publishes the next buffer to all waves.
    wait_async_zero();
    __syncthreads();
  }

  // ---- epilogue: C/D layout -> VGPR r holds M = r (lanes 0-15) / 8+r (16-31) ----
#pragma unroll
  for (int i = 0; i < 4; ++i) {
    const int mbase = tm + wm + i * 16 + hi * 8;
#pragma unroll
    for (int j = 0; j < 2; ++j) {
      const int ncol = tn + wn + j * 16 + l15;
#pragma unroll
      for (int r = 0; r < 8; ++r)
        z[(size_t)(mbase + r) * NN + ncol] = acc[i][j][r];
    }
  }
}

// ---------------------------------------------------------------------------
// Gate / state update: i,f,o sigmoid; g relu; c = f*c + i*g; h = o*relu(c)
// ---------------------------------------------------------------------------
__global__ __launch_bounds__(256) void lstm_gates_kernel(
    const float* __restrict__ z, const float* __restrict__ bias,
    float* __restrict__ c, unsigned short* __restrict__ h) {
  size_t i = (size_t)blockIdx.x * 256 + threadIdx.x;
  if (i >= (size_t)BB * HH) return;
  const int b  = (int)(i >> 10);
  const int hh = (int)(i & 1023);
  const float* zr = z + (size_t)b * NN;
  const float zi = zr[hh]            + bias[hh];
  const float zf = zr[HH + hh]       + bias[HH + hh];
  const float zg = zr[2 * HH + hh]   + bias[2 * HH + hh];
  const float zo = zr[3 * HH + hh]   + bias[3 * HH + hh];
  const float si = 1.0f / (1.0f + __expf(-zi));
  const float sf = 1.0f / (1.0f + __expf(-zf));
  const float g  = fmaxf(zg, 0.0f);
  const float so = 1.0f / (1.0f + __expf(-zo));
  const float cn = sf * c[i] + si * g;
  c[i] = cn;
  h[i] = f2bf(so * fmaxf(cn, 0.0f));
}

// ---------------------------------------------------------------------------
// Dense head: out[b][o] = sum_k h[b][k]*Wd[k][o] + bd[o]   (0.75 GFLOP, minor)
// ---------------------------------------------------------------------------
__global__ __launch_bounds__(256) void dense_head_kernel(
    const unsigned short* __restrict__ h, const float* __restrict__ Wd,
    const float* __restrict__ bd, float* __restrict__ out) {
  size_t i = (size_t)blockIdx.x * 256 + threadIdx.x;
  if (i >= (size_t)BB * OO) return;
  const int b = (int)(i / OO);
  const int o = (int)(i % OO);
  const unsigned short* hr = h + (size_t)b * HH;
  float s = bd[o];
#pragma unroll 4
  for (int k = 0; k < HH; ++k) s += bf2f(hr[k]) * Wd[(size_t)k * OO + o];
  out[i] = s;
}

static inline size_t alup(size_t x) { return (x + 255) & ~(size_t)255; }

extern "C" void kernel_launch(void* const* d_in, const int* in_sizes, int n_in,
                              void* d_out, int out_size, void* d_ws, size_t ws_size,
                              hipStream_t stream) {
  const float* x    = (const float*)d_in[0];  // [B,T,F]
  const float* Wk   = (const float*)d_in[1];  // [F,4H]
  const float* Wr   = (const float*)d_in[2];  // [H,4H]
  const float* bias = (const float*)d_in[3];  // [4H]
  const float* Wd   = (const float*)d_in[4];  // [H,O]
  const float* bd   = (const float*)d_in[5];  // [O]
  float* out = (float*)d_out;

  char* ws = (char*)d_ws;
  size_t off = 0;
  unsigned short* wct = (unsigned short*)(ws + off); off += alup((size_t)NN * KPAD * 2);
  unsigned short* hbf = (unsigned short*)(ws + off); off += alup((size_t)BB * HH * 2);
  float*          cst = (float*)(ws + off);          off += alup((size_t)BB * HH * 4);
  float*          zbf = (float*)(ws + off);          off += alup((size_t)BB * NN * 4);
  (void)ws_size; (void)in_sizes; (void)n_in; (void)out_size;

  // One-time (per launch) weight pack + state init
  {
    size_t n = (size_t)NN * KPAD;
    build_wct_kernel<<<dim3((unsigned)((n + 255) / 256)), 256, 0, stream>>>(Wk, Wr, wct);
  }
  {
    size_t n = (size_t)BB * HH;
    init_state_kernel<<<dim3((unsigned)((n + 255) / 256)), 256, 0, stream>>>(hbf, cst);
  }

  // 64 serial recurrence steps: WMMA GEMM + gate update
  dim3 ggrid(BB / 128, NN / 128);
  unsigned gblocks = (unsigned)(((size_t)BB * HH + 255) / 256);
  for (int t = 0; t < TT; ++t) {
    lstm_gemm_step_kernel<<<ggrid, 256, 0, stream>>>(x, wct, hbf, zbf, t);
    lstm_gates_kernel<<<gblocks, 256, 0, stream>>>(zbf, bias, cst, hbf);
  }

  // Dense head on last h
  {
    size_t n = (size_t)BB * OO;
    dense_head_kernel<<<dim3((unsigned)((n + 255) / 256)), 256, 0, stream>>>(hbf, Wd, bd, out);
  }
}
